// Attention_16441134809293
// MI455X (gfx1250) — matrix operand order repaired and checked by reading, not compiled, not run on hardware
//
#include <hip/hip_runtime.h>
#include <hip/hip_bf16.h>

// ---------------------------------------------------------------------------
// Problem constants (match reference)
// ---------------------------------------------------------------------------
#define TT   2048
#define DD   4096
#define NN   32
#define KK   8
#define HH   128
#define NH   4096       // N * H
#define KH   1024       // K * H
#define NEG_INF_F (-1e9f)

typedef unsigned short u16;
typedef __attribute__((ext_vector_type(8)))  float        v8f;
typedef __attribute__((ext_vector_type(16))) __bf16       v16bf;
typedef __attribute__((ext_vector_type(4)))  unsigned int u32x4;
typedef __attribute__((ext_vector_type(4)))  int          i32x4;
typedef __attribute__((ext_vector_type(8)))  int          i32x8;

union BF16Frag { v16bf v; u32x4 q[2]; };

__device__ __forceinline__ u16 f32_to_bf16(float f) {
    unsigned int u = __float_as_uint(f);
    u += 0x7FFFu + ((u >> 16) & 1u);          // round-to-nearest-even
    return (u16)(u >> 16);
}

__device__ __forceinline__ v16bf load_frag2(const u16* p0, const u16* p1) {
    BF16Frag u;
    u.q[0] = *(const u32x4*)p0;
    u.q[1] = *(const u32x4*)p1;
    return u.v;
}

__device__ __forceinline__ v16bf load_frag_k(const u16* p, int stride) {
    v16bf r;
#pragma unroll
    for (int s = 0; s < 16; ++s)
        r[s] = __builtin_bit_cast(__bf16, p[(size_t)((s & 7) + ((s >> 3) << 4)) * stride]);
    return r;
}

__device__ __forceinline__ v8f wmma_bf16(v16bf a, v16bf b, v8f c) {
    return __builtin_amdgcn_wmma_f32_16x16x32_bf16(
        false, a, false, b, (short)0, c, false, false);
}

// ---------------------------------------------------------------------------
// Tensor Data Mover: 2D bf16 tile (tile_d0 contiguous elems x tile_d1 rows,
// row stride = stride_elems) from global -> LDS (packed row-major).
// D# packing per CDNA5 ISA (groups 2/3 = 0 for 2D).
// This toolchain exposes the 6-arg builtin:
//   (u32x4 g0, i32x8 g1, i32x4 g2, i32x4 g3, i32x8, i32 cpol)
// ---------------------------------------------------------------------------
#if __has_builtin(__builtin_amdgcn_tensor_load_to_lds)
#define HAVE_TDM 1
__device__ __forceinline__ void tdm_load_2d(void* ldsDst, const void* gsrc,
                                            unsigned tensor_d0, unsigned tensor_d1,
                                            unsigned tile_d0, unsigned tile_d1,
                                            unsigned long long stride_elems)
{
    const unsigned long long ga = (unsigned long long)(uintptr_t)gsrc;
    u32x4 g0;
    g0.x = 1u;                                   // count=1, user mode
    g0.y = (unsigned)(uintptr_t)ldsDst;          // LDS byte offset (addr[31:0])
    g0.z = (unsigned)ga;                         // global_addr[31:0]
    g0.w = (unsigned)(ga >> 32) | 0x80000000u;   // global_addr[56:32] | type=2
    i32x8 g1;
    g1[0] = 0x00010000;                          // workgroup_mask=0, data_size=1 (2B)
    g1[1] = (int)((tensor_d0 & 0xFFFFu) << 16);  // tensor_dim0[15:0]
    g1[2] = (int)((tensor_d0 >> 16) | ((tensor_d1 & 0xFFFFu) << 16));
    g1[3] = (int)((tensor_d1 >> 16) | (tile_d0 << 16));
    g1[4] = (int)(tile_d1 & 0xFFFFu);            // tile_dim1 (tile_dim2=0)
    g1[5] = (int)(unsigned)(stride_elems & 0xFFFFFFFFull);   // dim0_stride[31:0]
    g1[6] = (int)(unsigned)(stride_elems >> 32);             // dim0_stride[47:32]
    g1[7] = 0;
    const i32x4 z4 = {0, 0, 0, 0};
    const i32x8 z8 = {0, 0, 0, 0, 0, 0, 0, 0};
    __builtin_amdgcn_tensor_load_to_lds(g0, g1, z4, z4, z8, 0);
}
#else
#define HAVE_TDM 0
#endif

// ---------------------------------------------------------------------------
// fp32 -> bf16 bulk convert (8 elems / thread, 2048 / block)
// ---------------------------------------------------------------------------
__global__ __launch_bounds__(256, 1)
void convert_bf16_kernel(const float* __restrict__ in, u16* __restrict__ out)
{
    const size_t base = (size_t)blockIdx.x * 2048 + (size_t)threadIdx.x * 8;
    float4 a = *(const float4*)(in + base);
    float4 b = *(const float4*)(in + base + 4);
    u16 r[8];
    r[0] = f32_to_bf16(a.x); r[1] = f32_to_bf16(a.y);
    r[2] = f32_to_bf16(a.z); r[3] = f32_to_bf16(a.w);
    r[4] = f32_to_bf16(b.x); r[5] = f32_to_bf16(b.y);
    r[6] = f32_to_bf16(b.z); r[7] = f32_to_bf16(b.w);
    *(u32x4*)(out + base) = *(u32x4*)r;
}

// ---------------------------------------------------------------------------
// GEMM: C[M x Ncols](f32) = A[M x Kdim](bf16) * B(bf16), WMMA bf16 math.
// BM=BN=128, BK=32, 256 threads (8 waves). TDM double-buffered LDS staging.
// wmode==0: B row-major, (k, j) at B[k*ldb + j]
// wmode==1: per-head weight [head][Kdim][128]; column-block == head.
// ---------------------------------------------------------------------------
__global__ __launch_bounds__(256, 1)
void gemm_bf16_kernel(const u16* __restrict__ A, int lda,
                      const u16* __restrict__ B, int ldb, int wmode,
                      int Kdim, float* __restrict__ C, int ldc)
{
    __shared__ u16 As[2][128 * 32];
    __shared__ u16 Bs[2][32 * 128];

    const int tid  = threadIdx.x;
    const int lane = tid & 31;
    const int wave = tid >> 5;
    const int wr   = wave & 3;       // 4 row groups of 32
    const int wc   = wave >> 2;      // 2 col groups of 64
    const int brow = blockIdx.y * 128;
    const int bcol = blockIdx.x * 128;

    const u16* Bbase = wmode ? (B + (size_t)blockIdx.x * (size_t)Kdim * 128)
                             : (B + bcol);
    const int bstride = wmode ? 128 : ldb;
    const int nIter = Kdim >> 5;

    v8f acc[2][4];
#pragma unroll
    for (int mt = 0; mt < 2; ++mt)
#pragma unroll
        for (int nt = 0; nt < 4; ++nt) acc[mt][nt] = (v8f)0.0f;

    const int half = lane >> 4;
    const int nc   = lane & 15;
    const int c0   = half * 8;

#if HAVE_TDM
    if (wave == 0) {    // prologue: buffer 0 in flight
        tdm_load_2d(&As[0][0], A + (size_t)brow * lda, (unsigned)lda, 1u << 20,
                    32u, 128u, (unsigned long long)lda);
        tdm_load_2d(&Bs[0][0], Bbase, (unsigned)bstride, 1u << 20,
                    128u, 32u, (unsigned long long)bstride);
    }
#endif

    for (int it = 0; it < nIter; ++it) {
        const int buf = it & 1;
#if HAVE_TDM
        if (wave == 0) {
            if (it + 1 < nIter) {   // prefetch next pair into other buffer
                const int k0n = (it + 1) * 32;
                const int nb  = buf ^ 1;
                tdm_load_2d(&As[nb][0], A + (size_t)brow * lda + k0n,
                            (unsigned)lda, 1u << 20, 32u, 128u,
                            (unsigned long long)lda);
                tdm_load_2d(&Bs[nb][0], Bbase + (size_t)k0n * bstride,
                            (unsigned)bstride, 1u << 20, 128u, 32u,
                            (unsigned long long)bstride);
                __builtin_amdgcn_s_wait_tensorcnt((unsigned short)2); // cur pair done
            } else {
                __builtin_amdgcn_s_wait_tensorcnt((unsigned short)0);
            }
        }
        __syncthreads();
#else
        // fallback: cooperative copy (16 u16 per thread per tile)
        {
            const int k0 = it * 32;
#pragma unroll
            for (int i = 0; i < 16; ++i) {
                const int ai = i * 256 + tid;
                As[buf][(ai >> 5) * 32 + (ai & 31)] =
                    A[(size_t)(brow + (ai >> 5)) * lda + (k0 + (ai & 31))];
                const int bi = i * 256 + tid;
                Bs[buf][(bi >> 7) * 128 + (bi & 127)] =
                    Bbase[(size_t)(k0 + (bi >> 7)) * bstride + (bi & 127)];
            }
        }
        __syncthreads();
#endif

        v16bf af[2], bfr[4];
#pragma unroll
        for (int mt = 0; mt < 2; ++mt) {
            const u16* p = &As[buf][(wr * 32 + mt * 16 + nc) * 32 + c0];
            af[mt] = load_frag2(p, p + 16);   // A 16x32: K runs {c0..+7, c0+16..+23}
        }
#pragma unroll
        for (int nt = 0; nt < 4; ++nt) {
            const u16* p = &Bs[buf][c0 * 128 + wc * 64 + nt * 16 + nc];
            bfr[nt] = load_frag_k(p, 128);   // B 32x16: lane == N col, K runs {c0..+7, c0+16..+23}
        }
#pragma unroll
        for (int mt = 0; mt < 2; ++mt)
#pragma unroll
            for (int nt = 0; nt < 4; ++nt)
                acc[mt][nt] = wmma_bf16(af[mt], bfr[nt], acc[mt][nt]);
        __syncthreads();   // all reads done before TDM overwrites this buffer
    }

#pragma unroll
    for (int mt = 0; mt < 2; ++mt)
#pragma unroll
        for (int nt = 0; nt < 4; ++nt)
#pragma unroll
            for (int v = 0; v < 8; ++v) {
                const int row = brow + wr * 32 + mt * 16 + v + 8 * half;
                const int col = bcol + wc * 64 + nt * 16 + nc;
                C[(size_t)row * ldc + col] = acc[mt][nt][v];
            }
}

// ---------------------------------------------------------------------------
// RMSNorm (+optional (1+scale)) (+optional RoPE), fp32 -> bf16.
// One wave per (t, head) row of H=128. in layout: [(t*heads + n)][128].
// out_kT==0: out[(n*T + t)*H + h]   (Q, V)
// out_kT==1: out[(n*H + h)*T + t]   (K transposed for logits B-fragments)
// ---------------------------------------------------------------------------
__global__ __launch_bounds__(128, 1)
void norm_rope_kernel(const float* __restrict__ in, int heads,
                      const float* __restrict__ scale, int do_rope,
                      int out_kT, u16* __restrict__ out)
{
    const int wave = threadIdx.x >> 5;
    const int lane = threadIdx.x & 31;
    const int r = blockIdx.x * 4 + wave;
    const int t = r / heads;
    const int n = r % heads;

    const float* row = in + (size_t)r * HH;
    const int i0 = lane * 2, i1 = lane * 2 + 1;          // 0..63

    float a0 = row[i0], a1 = row[i1];
    float b0 = row[i0 + 64], b1 = row[i1 + 64];

    float ss = a0 * a0 + a1 * a1 + b0 * b0 + b1 * b1;
#pragma unroll
    for (int m = 16; m >= 1; m >>= 1) ss += __shfl_xor(ss, m, 32);
    const float rn = rsqrtf(ss * (1.0f / (float)HH) + 1e-6f);
    a0 *= rn; a1 *= rn; b0 *= rn; b1 *= rn;

    if (scale) {
        a0 *= 1.0f + scale[i0];       a1 *= 1.0f + scale[i1];
        b0 *= 1.0f + scale[i0 + 64];  b1 *= 1.0f + scale[i1 + 64];
    }

    float o0 = a0, o1 = a1, o2 = b0, o3 = b1;
    if (do_rope) {
        const float pos = (float)t;
        const float ts0 = powf(10000.0f, (float)i0 * (1.0f / 64.0f));
        const float ts1 = powf(10000.0f, (float)i1 * (1.0f / 64.0f));
        const float an0 = pos / ts0, an1 = pos / ts1;
        const float s0 = sinf(an0), cc0 = cosf(an0);
        const float s1 = sinf(an1), cc1 = cosf(an1);
        o0 = a0 * cc0 - b0 * s0;  o2 = b0 * cc0 + a0 * s0;
        o1 = a1 * cc1 - b1 * s1;  o3 = b1 * cc1 + a1 * s1;
    }

    if (out_kT) {
        u16* base = out + (size_t)n * HH * TT;
        base[(size_t)i0 * TT + t]        = f32_to_bf16(o0);
        base[(size_t)i1 * TT + t]        = f32_to_bf16(o1);
        base[(size_t)(i0 + 64) * TT + t] = f32_to_bf16(o2);
        base[(size_t)(i1 + 64) * TT + t] = f32_to_bf16(o3);
    } else {
        u16* base = out + ((size_t)n * TT + t) * HH;
        base[i0]      = f32_to_bf16(o0);
        base[i1]      = f32_to_bf16(o1);
        base[i0 + 64] = f32_to_bf16(o2);
        base[i1 + 64] = f32_to_bf16(o3);
    }
}

// ---------------------------------------------------------------------------
// Flash attention (causal, GQA). One wave per (head n, 16-row Q tile).
// qb: [n][T][H] bf16, kbT: [kh][H][T] bf16, vb: [kh][T][H] bf16.
// aob: [T][NH] bf16 (consumed by output projection GEMM).
// ---------------------------------------------------------------------------
__global__ __launch_bounds__(128, 1)
void attn_kernel(const u16* __restrict__ qb, const u16* __restrict__ kbT,
                 const u16* __restrict__ vb, u16* __restrict__ aob)
{
    __shared__ u16 Pbuf[4][16 * 32];   // per-wave C->A re-layout staging

    const int wave = threadIdx.x >> 5;
    const int lane = threadIdx.x & 31;
    const int w    = blockIdx.x * 4 + wave;
    const int n    = w >> 7;           // / (T/16 = 128)
    const int qt   = w & 127;
    const int nk   = n >> 2;           // kv head (G = 4)
    const int qbase = qt * 16;

    const int half = lane >> 4;
    const int nc   = lane & 15;
    const int c0   = half * 8;

    v16bf qa[4];
    const u16* qrow = qb + ((size_t)n * TT + qbase + nc) * HH;
#pragma unroll
    for (int hc = 0; hc < 4; ++hc) {
        const u16* p = qrow + hc * 32 + c0;
        qa[hc] = load_frag2(p, p + 16);
    }

    float mrow[8], lrow[8];
#pragma unroll
    for (int v = 0; v < 8; ++v) { mrow[v] = -3.0e38f; lrow[v] = 0.0f; }
    v8f O[8];
#pragma unroll
    for (int h = 0; h < 8; ++h) O[h] = (v8f)0.0f;

    u16* Pw = &Pbuf[wave][0];

    for (int s0 = 0; s0 <= qbase + 15; s0 += 32) {
        v8f S0 = (v8f)0.0f, S1 = (v8f)0.0f;
#pragma unroll
        for (int hc = 0; hc < 4; ++hc) {
            const u16* kp = kbT + ((size_t)nk * HH + hc * 32 + c0) * TT + s0 + nc;
            v16bf b0 = load_frag_k(kp, TT);
            v16bf b1 = load_frag_k(kp + 16, TT);
            S0 = wmma_bf16(qa[hc], b0, S0);
            S1 = wmma_bf16(qa[hc], b1, S1);
        }

        const int col0 = s0 + nc, col1 = col0 + 16;
        float alpha[8];
#pragma unroll
        for (int v = 0; v < 8; ++v) {
            const int rowg = qbase + v + half * 8;
            const float s0v = S0[v] + (col0 > rowg ? NEG_INF_F : 0.0f);
            const float s1v = S1[v] + (col1 > rowg ? NEG_INF_F : 0.0f);
            float cm = fmaxf(s0v, s1v);
#pragma unroll
            for (int mm = 8; mm >= 1; mm >>= 1)
                cm = fmaxf(cm, __shfl_xor(cm, mm, 32));   // within 16-lane half
            const float mn = fmaxf(mrow[v], cm);
            alpha[v] = expf(mrow[v] - mn);
            mrow[v] = mn;
            const float p0 = expf(s0v - mn);
            const float p1 = expf(s1v - mn);
            float rs = p0 + p1;
#pragma unroll
            for (int mm = 8; mm >= 1; mm >>= 1)
                rs += __shfl_xor(rs, mm, 32);
            lrow[v] = lrow[v] * alpha[v] + rs;
            const int rl = v + half * 8;
            Pw[rl * 32 + nc]      = f32_to_bf16(p0);
            Pw[rl * 32 + nc + 16] = f32_to_bf16(p1);
        }
#pragma unroll
        for (int h = 0; h < 8; ++h)
#pragma unroll
            for (int v = 0; v < 8; ++v) O[h][v] *= alpha[v];

        asm volatile("s_wait_dscnt 0" ::: "memory");  // wave-local LDS RAW

        const u16* pp = Pw + nc * 32 + c0;
        v16bf pa = load_frag2(pp, pp + 16);
#pragma unroll
        for (int hc2 = 0; hc2 < 8; ++hc2) {
            const u16* vp = vb + ((size_t)nk * TT + s0 + c0) * HH + hc2 * 16 + nc;
            v16bf bv = load_frag_k(vp, HH);
            O[hc2] = wmma_bf16(pa, bv, O[hc2]);
        }
    }

    float inv[8];
#pragma unroll
    for (int v = 0; v < 8; ++v) inv[v] = 1.0f / lrow[v];
#pragma unroll
    for (int hc2 = 0; hc2 < 8; ++hc2)
#pragma unroll
        for (int v = 0; v < 8; ++v) {
            const int trow = qbase + v + half * 8;
            aob[(size_t)trow * NH + n * HH + hc2 * 16 + nc] =
                f32_to_bf16(O[hc2][v] * inv[v]);
        }
}

// ---------------------------------------------------------------------------
// Host-side orchestration
// ---------------------------------------------------------------------------
extern "C" void kernel_launch(void* const* d_in, const int* in_sizes, int n_in,
                              void* d_out, int out_size, void* d_ws, size_t ws_size,
                              hipStream_t stream)
{
    const float* x       = (const float*)d_in[0];
    // d_in[1] segment_pos (arange, recomputed), d_in[2] attn_mask (causal, built in)
    const float* q_w     = (const float*)d_in[3];   // [N][D][H]
    const float* kv_w    = (const float*)d_in[4];   // [2][K][D][H]
    const float* o_w     = (const float*)d_in[5];   // [N][H][D]
    const float* q_scale = (const float*)d_in[6];
    const float* k_scale = (const float*)d_in[7];
    float* out = (float*)d_out;                     // [T][D]

    const size_t MB = 1024 * 1024;
    char* ws = (char*)d_ws;
    u16*   xb   = (u16*)  (ws);                 // [T][D]      16 MB
    u16*   wqb  = (u16*)  (ws + 16  * MB);      // [N][D][H]   32 MB
    u16*   wkvb = (u16*)  (ws + 48  * MB);      // [2][K][D][H]16 MB
    u16*   owb  = (u16*)  (ws + 64  * MB);      // [NH][D]     32 MB
    float* q32  = (float*)(ws + 96  * MB);      // [T][NH]     32 MB
    float* k32  = (float*)(ws + 128 * MB);      // [T][KH]      8 MB
    float* v32  = (float*)(ws + 136 * MB);      // [T][KH]      8 MB
    u16*   qb   = (u16*)  (ws + 144 * MB);      // [N][T][H]   16 MB
    u16*   kbT  = (u16*)  (ws + 160 * MB);      // [K][H][T]    4 MB
    u16*   vb   = (u16*)  (ws + 164 * MB);      // [K][T][H]    4 MB
    u16*   aob  = (u16*)  (ws + 96  * MB);      // aliases q32 (free after norm)

    // ---- bf16 pre-conversion (activations + weights) ----
    convert_bf16_kernel<<<(TT * DD) / 2048, 256, 0, stream>>>(x, xb);
    convert_bf16_kernel<<<(NN * DD * HH) / 2048, 256, 0, stream>>>(q_w, wqb);
    convert_bf16_kernel<<<(2 * KK * DD * HH) / 2048, 256, 0, stream>>>(kv_w, wkvb);
    convert_bf16_kernel<<<(NH * DD) / 2048, 256, 0, stream>>>(o_w, owb);

    // ---- projections: q = x @ q_w, k = x @ kv_w[0], v = x @ kv_w[1] ----
    gemm_bf16_kernel<<<dim3(NH / 128, TT / 128), 256, 0, stream>>>(
        xb, DD, wqb, 0, 1, DD, q32, NH);
    gemm_bf16_kernel<<<dim3(KH / 128, TT / 128), 256, 0, stream>>>(
        xb, DD, wkvb, 0, 1, DD, k32, KH);
    gemm_bf16_kernel<<<dim3(KH / 128, TT / 128), 256, 0, stream>>>(
        xb, DD, wkvb + (size_t)KK * DD * HH, 0, 1, DD, v32, KH);

    // ---- RMSNorm (+scale) (+RoPE), bf16 repack ----
    norm_rope_kernel<<<(TT * NN) / 4, 128, 0, stream>>>(q32, NN, q_scale, 1, 0, qb);
    norm_rope_kernel<<<(TT * KK) / 4, 128, 0, stream>>>(k32, KK, k_scale, 1, 1, kbT);
    norm_rope_kernel<<<(TT * KK) / 4, 128, 0, stream>>>(v32, KK, nullptr, 0, 0, vb);

    // ---- flash attention (writes bf16) ----
    attn_kernel<<<(NN * (TT / 16)) / 4, 128, 0, stream>>>(qb, kbT, vb, aob);

    // ---- output projection: out = ao @ o_w ----
    gemm_bf16_kernel<<<dim3(DD / 128, TT / 128), 256, 0, stream>>>(
        aob, NH, owb, DD, 0, NH, out, DD);
}
